// DecoderGNN_24962349924439
// MI455X (gfx1250) — matrix operand order repaired and checked by reading, not compiled, not run on hardware
//
#include <hip/hip_runtime.h>
#include <hip/hip_bf16.h>
#include <math.h>

typedef __attribute__((ext_vector_type(16))) __bf16 v16bf;
typedef __attribute__((ext_vector_type(8)))  float  v8f;

// ---------------------------------------------------------------------------
// CDNA5 async global->LDS copy (ASYNCcnt-tracked), via inline asm.
// LDS destination = low 32 bits of the generic pointer (aperture mapping
// truncates generic LDS addresses to addr[31:0]).
// ---------------------------------------------------------------------------
__device__ __forceinline__ void async_load_b128(const void* gsrc, void* ldst) {
  unsigned lofs = (unsigned)(uintptr_t)ldst;
  asm volatile("global_load_async_to_lds_b128 %0, %1, off"
               :: "v"(lofs), "v"(gsrc) : "memory");
}
__device__ __forceinline__ void wait_async0() {
  asm volatile("s_wait_asynccnt 0" ::: "memory");
}

__device__ __forceinline__ v8f wmma_bf16(v16bf a, v16bf b, v8f c) {
  // D = A(16x32 bf16) x B(32x16 bf16) + C(16x16 f32)
  return __builtin_amdgcn_wmma_f32_16x16x32_bf16(
      /*neg_a=*/false, a, /*neg_b=*/false, b,
      /*c_mod=*/(short)0, c, /*reuse_a=*/false, /*reuse_b=*/false);
}

// Fast transcendentals on the scan critical path (native V_EXP/V_RCP/V_TANH).
__device__ __forceinline__ float fast_sigmoid(float x) {
  float e = __builtin_amdgcn_exp2f(-1.442695041f * x);   // exp(-x)
  return __builtin_amdgcn_rcpf(1.f + e);
}
__device__ __forceinline__ float fast_tanh(float x) {
#if __has_builtin(__builtin_amdgcn_tanhf)
  return __builtin_amdgcn_tanhf(x);
#else
  float e = __builtin_amdgcn_exp2f(2.885390082f * x);    // exp(2x)
  return 1.f - 2.f * __builtin_amdgcn_rcpf(e + 1.f);
#endif
}

// ---------------------------------------------------------------------------
// One-shot packers (weights are tiny; all heavy loops read prepacked bf16)
// ---------------------------------------------------------------------------
__global__ void to_bf16(const float* __restrict__ src, __bf16* __restrict__ dst, int n)
{
  int i = blockIdx.x * blockDim.x + threadIdx.x;
  if (i < n) dst[i] = (__bf16)src[i];
}

// Bp layout: [(nt*(K/32)+ks)*32 + lane][16] ; lane holds col nt*16+(lane&15),
// K = ks*32 + (lane>>4)*16 + i.  bt==1 means B stored NxK (use B^T).
__global__ void pack_b(const float* __restrict__ B, __bf16* __restrict__ Bp,
                       int K, int N, int bt)
{
  int idx = blockIdx.x * blockDim.x + threadIdx.x;
  int total = (N >> 4) * (K >> 5) * 32;
  if (idx >= total) return;
  int lane = idx & 31;
  int Ks = K >> 5;
  int nt = (idx >> 5) / Ks;
  int ks = (idx >> 5) % Ks;
  int n = nt * 16 + (lane & 15);
  int kb = ks * 32 + (lane >> 4) * 16;
  __bf16* dst = Bp + (size_t)idx * 16;
  #pragma unroll
  for (int i = 0; i < 16; ++i) {
    int k = kb + i;
    float v = bt ? B[(size_t)n * K + k] : B[(size_t)k * N + n];
    dst[i] = (__bf16)v;
  }
}

// ---------------------------------------------------------------------------
// bf16 WMMA GEMM, prepacked B, 16x64 tile per wave (4 accumulators).
// A: bf16 MxK row-major.  Output f32 or bf16 (template).
// ---------------------------------------------------------------------------
template<bool OUTBF>
__global__ __launch_bounds__(256)
void gemm_bf16p(const __bf16* __restrict__ A, const __bf16* __restrict__ Bp,
                const float* __restrict__ bias0, const float* __restrict__ bias1,
                float* __restrict__ Cf, __bf16* __restrict__ Cb,
                int M, int N, int K)
{
  const int wave = blockIdx.x * 8 + (threadIdx.x >> 5);
  const int lane = threadIdx.x & 31;
  const int ngrp = N >> 6;                    // 64-col groups
  const int mt = wave / ngrp;
  const int ng = wave % ngrp;
  if (mt * 16 >= M) return;
  const int lo = lane & 15, hi = lane >> 4;
  const int arow = mt * 16 + lo;
  const int Ks = K >> 5;

  v8f acc[4] = {{}, {}, {}, {}};
  for (int ks = 0; ks < Ks; ++ks) {
    v16bf af;
    const __bf16* ap = A + (size_t)arow * K + ks * 32 + hi * 16;
    #pragma unroll
    for (int i = 0; i < 16; ++i) af[i] = ap[i];
    #pragma unroll
    for (int u = 0; u < 4; ++u) {
      const __bf16* bp = Bp + ((size_t)((ng * 4 + u) * Ks + ks) * 32 + lane) * 16;
      v16bf bf;
      #pragma unroll
      for (int i = 0; i < 16; ++i) bf[i] = bp[i];
      acc[u] = wmma_bf16(af, bf, acc[u]);
    }
  }
  #pragma unroll
  for (int u = 0; u < 4; ++u) {
    const int ccol = (ng * 4 + u) * 16 + lo;
    float bv = 0.f;
    if (bias0) bv += bias0[ccol];
    if (bias1) bv += bias1[ccol];
    const int crow = mt * 16 + 8 * hi;
    #pragma unroll
    for (int r = 0; r < 8; ++r) {
      int rr = crow + r;
      if (rr < M) {
        float v = acc[u][r] + bv;
        if (OUTBF) Cb[(size_t)rr * N + ccol] = (__bf16)v;
        else       Cf[(size_t)rr * N + ccol] = v;
      }
    }
  }
}

// ---------------------------------------------------------------------------
// Persistent single-workgroup LSTM scan over 4096 nodes (serial dependence).
// 32 waves; wave w keeps its 128x16 slice of w_hh^T in registers as WMMA
// B-fragments.  h: LDS bf16 (16x128, rows 8..15 zero).  c: registers.
// GX[n] rows double-buffered into LDS with GLOBAL_LOAD_ASYNC_TO_LDS_B128.
// gbuf and g_lds carry 16 rows (8 real + 8 zero) so the seed and the gate
// publish are branch-free (no exec-mask toggles on the critical path).
// ---------------------------------------------------------------------------
__global__ __launch_bounds__(1024)
void lstm_scan(const float* __restrict__ GX, const float* __restrict__ w_hh,
               const float* __restrict__ hidden, const float* __restrict__ cell,
               __bf16* __restrict__ outs16, float* __restrict__ hT,
               float* __restrict__ cT, int Nn)
{
  __shared__ __bf16 h_bf[16 * 128];          //  4KB
  __shared__ float  g_lds[16 * 512];         // 32KB (rows 8..15 written, unread)
  __shared__ float  gbuf[2][16 * 512];       // 64KB (rows 8..15 zero)

  const int tid  = threadIdx.x;
  const int lane = tid & 31;
  const int wv   = tid >> 5;
  const int lo   = lane & 15;
  const int hi   = lane >> 4;
  const int col  = wv * 16 + lo;             // gate column in [0,512)

  // Resident weight B-fragments: B[k][col] = w_hh[col*128 + k] (contig in k)
  v16bf wfrag[4];
  #pragma unroll
  for (int kf = 0; kf < 4; ++kf) {
    const float* wp = w_hh + (size_t)col * 128 + kf * 32 + hi * 16;
    #pragma unroll
    for (int i = 0; i < 16; ++i) wfrag[kf][i] = (__bf16)wp[i];
  }

  const int t = tid >> 7;                    // 0..7
  const int j = tid & 127;                   // 0..127
  float c_state = cell[t * 128 + j];
  float h_cur   = hidden[t * 128 + j];
  h_bf[t * 128 + j] = (__bf16)h_cur;         // rows 0..7
  h_bf[1024 + tid]  = (__bf16)0.f;           // zero pad rows 8..15
  // zero pad rows 8..15 of both gbuf buffers (read by lanes 16..31 forever)
  *(float4*)&gbuf[0][4096 + tid * 4] = float4{0.f, 0.f, 0.f, 0.f};
  *(float4*)&gbuf[1][4096 + tid * 4] = float4{0.f, 0.f, 0.f, 0.f};

  // Preload GX[0]: 1024 threads x 16B = 16KB row
  async_load_b128(GX + tid * 4, &gbuf[0][tid * 4]);
  wait_async0();
  __syncthreads();

  const int mask = Nn - 1;                   // Nn = 4096 (pow2)
  for (int n = 0; n < Nn; ++n) {
    // kick next row while we compute this one (last iter reloads row 0)
    const int np = (n + 1) & mask;
    async_load_b128(GX + (size_t)np * 4096 + tid * 4,
                    &gbuf[(n + 1) & 1][tid * 4]);
    const float* gb = gbuf[n & 1];

    // Seed accumulator with GX[n]; rows 8..15 read the zero pad (branch-free)
    v8f acc;
    #pragma unroll
    for (int r = 0; r < 8; ++r) acc[r] = gb[(r + 8 * hi) * 512 + col];

    // A-fragments from h (LDS bf16), K = 4 x 32
    #pragma unroll
    for (int kf = 0; kf < 4; ++kf) {
      v16bf af;
      const __bf16* hp = h_bf + lo * 128 + kf * 32 + hi * 16;
      #pragma unroll
      for (int i = 0; i < 16; ++i) af[i] = hp[i];
      acc = wmma_bf16(af, wfrag[kf], acc);
    }

    // Publish gates; all lanes store (rows 8..15 are zeros, never read)
    #pragma unroll
    for (int r = 0; r < 8; ++r) g_lds[(r + 8 * hi) * 512 + col] = acc[r];
    __syncthreads();

    // Elementwise LSTM cell (torch gate order i, f, g, o)
    float ig = g_lds[t * 512 +       j];
    float fg = g_lds[t * 512 + 128 + j];
    float gg = g_lds[t * 512 + 256 + j];
    float og = g_lds[t * 512 + 384 + j];
    c_state = fast_sigmoid(fg) * c_state + fast_sigmoid(ig) * fast_tanh(gg);
    h_cur   = fast_sigmoid(og) * fast_tanh(c_state);
    h_bf[t * 128 + j] = (__bf16)h_cur;
    outs16[(size_t)n * 1024 + t * 128 + j] = (__bf16)h_cur;

    wait_async0();        // next-row tile landed in LDS
    __syncthreads();
  }

  hT[t * 128 + j] = h_cur;
  cT[t * 128 + j] = c_state;
}

// ---------------------------------------------------------------------------
// GAT edge softmax pipeline (fs/fd are L2-resident: 16MB each)
// ---------------------------------------------------------------------------
__global__ void init_ws(float* agg, float* denom, unsigned* menc, int nAgg, int nNT)
{
  int i = blockIdx.x * blockDim.x + threadIdx.x;
  if (i < nAgg) agg[i] = 0.f;
  if (i < nNT) { denom[i] = 0.f; menc[i] = 0x007FFFFFu; /* enc(-inf) */ }
}

__global__ __launch_bounds__(256)
void edge_logits(const float* __restrict__ fs, const float* __restrict__ fd,
                 const float* __restrict__ attn, const int* __restrict__ src,
                 const int* __restrict__ dst, float* __restrict__ logits,
                 unsigned* __restrict__ menc, int E)
{
  const int e = blockIdx.x * 8 + (threadIdx.x >> 5);
  if (e >= E) return;
  const int lane = threadIdx.x & 31;
  const int s = src[e], d = dst[e];
  const int t = lane >> 2;     // 0..7
  const int q = lane & 3;      // quarter of H
  const float4* fsp = (const float4*)(fs + (size_t)s * 1024 + t * 128 + q * 32);
  const float4* fdp = (const float4*)(fd + (size_t)d * 1024 + t * 128 + q * 32);
  const float4* ap  = (const float4*)(attn + q * 32);
  float acc = 0.f;
  #pragma unroll
  for (int i = 0; i < 8; ++i) {
    float4 a4 = fsp[i], b4 = fdp[i], w4 = ap[i];
    float v;
    v = a4.x + b4.x; acc += w4.x * (v > 0.f ? v : 0.2f * v);
    v = a4.y + b4.y; acc += w4.y * (v > 0.f ? v : 0.2f * v);
    v = a4.z + b4.z; acc += w4.z * (v > 0.f ? v : 0.2f * v);
    v = a4.w + b4.w; acc += w4.w * (v > 0.f ? v : 0.2f * v);
  }
  acc += __shfl_xor(acc, 1);
  acc += __shfl_xor(acc, 2);
  if (q == 0) {
    logits[(size_t)e * 8 + t] = acc;
    unsigned u = __float_as_uint(acc);
    u = ((int)u >= 0) ? (u | 0x80000000u) : ~u;   // order-preserving encode
    atomicMax(&menc[d * 8 + t], u);
  }
}

__global__ void edge_expdenom(const int* __restrict__ dst, float* __restrict__ logits,
                              const unsigned* __restrict__ menc,
                              float* __restrict__ denom, int E8)
{
  int i = blockIdx.x * blockDim.x + threadIdx.x;
  if (i >= E8) return;
  int e = i >> 3, t = i & 7;
  int d = dst[e];
  unsigned u = menc[d * 8 + t];
  unsigned bits = (u & 0x80000000u) ? (u ^ 0x80000000u) : ~u;
  float m = __uint_as_float(bits);
  float a = __builtin_amdgcn_exp2f(1.442695041f * (logits[i] - m));
  logits[i] = a;
  atomicAdd(&denom[d * 8 + t], a);
}

__global__ __launch_bounds__(256)
void edge_aggregate(const float* __restrict__ fs, const int* __restrict__ src,
                    const int* __restrict__ dst, const float* __restrict__ logits,
                    const float* __restrict__ denom, float* __restrict__ agg, int E)
{
  const int e = blockIdx.x * 8 + (threadIdx.x >> 5);
  if (e >= E) return;
  const int lane = threadIdx.x & 31;
  const int s = src[e], d = dst[e];
  #pragma unroll
  for (int t = 0; t < 8; ++t) {
    float alpha = logits[(size_t)e * 8 + t] / denom[d * 8 + t];
    const float4 v = ((const float4*)(fs + (size_t)s * 1024 + t * 128))[lane];
    float* ag = agg + (size_t)d * 1024 + t * 128 + lane * 4;
    atomicAdd(ag + 0, alpha * v.x);
    atomicAdd(ag + 1, alpha * v.y);
    atomicAdd(ag + 2, alpha * v.z);
    atomicAdd(ag + 3, alpha * v.w);
  }
}

__global__ void out_head(const float* __restrict__ agg, const float* __restrict__ gat_b,
                         const float* __restrict__ out_w, const float* __restrict__ out_b,
                         float* __restrict__ y, int M)
{
  int m = blockIdx.x * blockDim.x + threadIdx.x;
  if (m >= M) return;
  const float4* a = (const float4*)(agg + (size_t)m * 128);
  const float4* g = (const float4*)gat_b;
  const float4* w = (const float4*)out_w;
  float acc = 0.f;
  #pragma unroll
  for (int jj = 0; jj < 32; ++jj) {
    float4 a4 = a[jj], g4 = g[jj], w4 = w[jj];
    acc += (a4.x + g4.x) * w4.x + (a4.y + g4.y) * w4.y +
           (a4.z + g4.z) * w4.z + (a4.w + g4.w) * w4.w;
  }
  y[m] = acc + out_b[0];
}

// ---------------------------------------------------------------------------
extern "C" void kernel_launch(void* const* d_in, const int* in_sizes, int n_in,
                              void* d_out, int out_size, void* d_ws, size_t ws_size,
                              hipStream_t stream)
{
  const float* input  = (const float*)d_in[0];   // (4096, 8, 64)
  const float* hidden = (const float*)d_in[1];   // (1, 8, 128)
  const float* cell   = (const float*)d_in[2];   // (1, 8, 128)
  const float* pre_w  = (const float*)d_in[3];   // (64, 128)
  const float* pre_b  = (const float*)d_in[4];   // (128)
  const float* w_ih   = (const float*)d_in[5];   // (512, 128)
  const float* w_hh   = (const float*)d_in[6];   // (512, 128)
  const float* b_ih   = (const float*)d_in[7];   // (512)
  const float* b_hh   = (const float*)d_in[8];   // (512)
  const float* gwsrc  = (const float*)d_in[9];   // (128, 128)
  const float* gwdst  = (const float*)d_in[10];  // (128, 128)
  const float* gattn  = (const float*)d_in[11];  // (128)
  const float* gat_b  = (const float*)d_in[12];  // (128)
  const float* out_w  = (const float*)d_in[13];  // (128, 1)
  const float* out_b  = (const float*)d_in[14];  // (1)
  const int*   esrc   = (const int*)d_in[15];
  const int*   edst   = (const int*)d_in[16];

  const int Nn = 4096, T = 8, H = 128, IN = 64;
  const int MT = Nn * T;                 // 32768
  const int E  = in_sizes[15];           // 69632
  const size_t MB = 1024 * 1024;

  // Workspace layout (bytes):
  //   [0,64MB)    GX (f32)      -> after scan reused: fs [0,16MB), fd [16,32MB)
  //   [64,68)     Xb   = input bf16
  //   [68,76)     X16  = pre-GEMM output bf16
  //   [76,84)     outs16 (bf16)
  //   [84,100)    agg (f32)
  //   [100,..)    logits, menc, denom, weight packs
  char* ws = (char*)d_ws;
  float*    GX     = (float*)(ws);
  float*    fs     = (float*)(ws);
  float*    fd     = (float*)(ws + 16 * MB);
  __bf16*   Xb     = (__bf16*)(ws + 64 * MB);
  __bf16*   X16    = (__bf16*)(ws + 68 * MB);
  __bf16*   outs16 = (__bf16*)(ws + 76 * MB);
  float*    agg    = (float*)(ws + 84 * MB);
  float*    logits = (float*)(ws + 100 * MB);
  unsigned* menc   = (unsigned*)((char*)logits + (size_t)E * 8 * 4);
  float*    denom  = (float*)((char*)menc + (size_t)Nn * 8 * 4);
  __bf16*   pw_p   = (__bf16*)((char*)denom + (size_t)Nn * 8 * 4);   // 16KB
  __bf16*   wih_p  = pw_p  + 64 * 128;                               // 128KB
  __bf16*   gws_p  = wih_p + 512 * 128;                              // 32KB
  __bf16*   gwd_p  = gws_p + 128 * 128;                              // 32KB

  float* y  = (float*)d_out;
  float* hT = y + MT;             // 32768
  float* cT = hT + T * H;         // +1024

  // 0) one-shot conversions / weight packing + scratch init
  to_bf16<<<(Nn * T * IN + 255) / 256, 256, 0, stream>>>(input, Xb, Nn * T * IN);
  pack_b<<<(8 * 2 * 32 + 255) / 256, 256, 0, stream>>>(pre_w, pw_p, IN, H, 0);
  pack_b<<<(32 * 4 * 32 + 255) / 256, 256, 0, stream>>>(w_ih, wih_p, H, 4 * H, 1);
  pack_b<<<(8 * 4 * 32 + 255) / 256, 256, 0, stream>>>(gwsrc, gws_p, H, H, 0);
  pack_b<<<(8 * 4 * 32 + 255) / 256, 256, 0, stream>>>(gwdst, gwd_p, H, H, 0);
  init_ws<<<(Nn * T * H + 255) / 256, 256, 0, stream>>>(agg, denom, menc,
                                                        Nn * T * H, Nn * T);

  // 1) X16 = bf16( input @ pre_w + pre_b )        (32768 x 64 x 128)
  gemm_bf16p<true><<<(MT / 16) * (H / 64) / 8, 256, 0, stream>>>(
      Xb, pw_p, pre_b, nullptr, nullptr, X16, MT, H, IN);

  // 2) GX = X16 @ w_ih^T + (b_ih + b_hh)          (32768 x 128 x 512)
  gemm_bf16p<false><<<(MT / 16) * (4 * H / 64) / 8, 256, 0, stream>>>(
      X16, wih_p, b_ih, b_hh, GX, nullptr, MT, 4 * H, H);

  // 3) sequential LSTM scan; writes outs16, hT, cT
  lstm_scan<<<1, 1024, 0, stream>>>(GX, w_hh, hidden, cell, outs16, hT, cT, Nn);

  // 4) fs/fd = outs @ gat_w{src,dst}              (32768 x 128 x 128, x2)
  gemm_bf16p<false><<<(MT / 16) * (H / 64) / 8, 256, 0, stream>>>(
      outs16, gws_p, nullptr, nullptr, fs, nullptr, MT, H, H);
  gemm_bf16p<false><<<(MT / 16) * (H / 64) / 8, 256, 0, stream>>>(
      outs16, gwd_p, nullptr, nullptr, fd, nullptr, MT, H, H);

  // 5) edge softmax: logits + segment-max, exp + segment-sum, weighted agg
  edge_logits<<<(E + 7) / 8, 256, 0, stream>>>(fs, fd, gattn, esrc, edst,
                                               logits, menc, E);
  edge_expdenom<<<(E * 8 + 255) / 256, 256, 0, stream>>>(edst, logits, menc,
                                                         denom, E * 8);
  edge_aggregate<<<(E + 7) / 8, 256, 0, stream>>>(fs, esrc, edst, logits,
                                                  denom, agg, E);

  // 6) y = (agg + gat_b) @ out_w + out_b
  out_head<<<(MT + 255) / 256, 256, 0, stream>>>(agg, gat_b, out_w, out_b, y, MT);
}
